// SlidingWindowAttention_5686536700429
// MI455X (gfx1250) — compile-verified, matrix operand order checked by reading, not run on hardware
//
#include <hip/hip_runtime.h>

typedef __attribute__((ext_vector_type(16))) __bf16 v16bf;
typedef __attribute__((ext_vector_type(8)))  float  v8f;
typedef __attribute__((ext_vector_type(4)))  int    v4i;
typedef __attribute__((address_space(1))) v4i* gv4i_p;   // global int4*
typedef __attribute__((address_space(3))) v4i* lv4i_p;   // LDS int4*

#define TT   4096
#define DD   2048
#define HH   32
#define DKH  64
#define BS   128
#define NB   (TT / BS)     // 32
#define BB   2
#define MTOT (BB * TT)     // 8192

#if __has_builtin(__builtin_amdgcn_global_load_async_to_lds_b128) && \
    __has_builtin(__builtin_amdgcn_s_wait_asynccnt)
#define USE_ASYNC_LDS 1
#else
#define USE_ASYNC_LDS 0
#endif

// ---------------------------------------------------------------------------
// fp32 -> bf16 elementwise cast (4 elems / thread)
// ---------------------------------------------------------------------------
__global__ void cast_bf16_kernel(const float* __restrict__ in,
                                 __bf16* __restrict__ out, int n4) {
  int i = blockIdx.x * blockDim.x + threadIdx.x;
  if (i < n4) {
    float4 f = reinterpret_cast<const float4*>(in)[i];
    union { __bf16 h[4]; uint2 u; } o;
    o.h[0] = (__bf16)f.x; o.h[1] = (__bf16)f.y;
    o.h[2] = (__bf16)f.z; o.h[3] = (__bf16)f.w;
    reinterpret_cast<uint2*>(out)[i] = o.u;
  }
}

// ---------------------------------------------------------------------------
// C[M,N] = A[M,K] * Bn[N,K]^T  (bf16 in, f32 accumulate via WMMA)
// Bn is the ORIGINAL weight layout [N,K] (k-contiguous), which is exactly the
// B-fragment layout WMMA wants -> both LDS tiles fill with b128 async copies.
// Block tile 128x128, BK=32, 8 waves (4 along M x 2 along N), wave tile 32x64.
// ---------------------------------------------------------------------------
__global__ __launch_bounds__(256) void gemm_bf16_kernel(
    const __bf16* __restrict__ A, const __bf16* __restrict__ Bn,
    __bf16* __restrict__ outB, float* __restrict__ outF,
    int M, int N, int K) {
  __shared__ __bf16 As[128 * 32];   // [row m][k]  k-contiguous
  __shared__ __bf16 Bs[128 * 32];   // [col n][k]  k-contiguous

  const int tid   = threadIdx.x;
  const int lane  = tid & 31;
  const int col   = lane & 15;
  const int rhalf = lane >> 4;
  const int kb    = rhalf * 8;      // ISA 16-bit A/B frag K split per half-wave
  const int wave  = tid >> 5;
  const int wm    = wave & 3;       // 4 waves along M
  const int wn    = wave >> 2;      // 2 waves along N
  const int m0    = blockIdx.y * 128;
  const int n0    = blockIdx.x * 128;

  const int trow = tid >> 1;            // 0..127 (m for A, n for B)
  const int tcol = (tid & 1) * 16;      // 0 / 16

  v8f acc[2][4];
#pragma unroll
  for (int mi = 0; mi < 2; mi++)
#pragma unroll
    for (int ni = 0; ni < 4; ni++) acc[mi][ni] = {};

  for (int k0 = 0; k0 < K; k0 += 32) {
    const __bf16* ga = A  + (size_t)(m0 + trow) * K + k0 + tcol;
    const __bf16* gb = Bn + (size_t)(n0 + trow) * K + k0 + tcol;

#if USE_ASYNC_LDS
    __builtin_amdgcn_global_load_async_to_lds_b128(
        (gv4i_p)ga, (lv4i_p)&As[trow * 32 + tcol], 0, 0);
    __builtin_amdgcn_global_load_async_to_lds_b128(
        (gv4i_p)ga, (lv4i_p)&As[trow * 32 + tcol], 16, 0);
    __builtin_amdgcn_global_load_async_to_lds_b128(
        (gv4i_p)gb, (lv4i_p)&Bs[trow * 32 + tcol], 0, 0);
    __builtin_amdgcn_global_load_async_to_lds_b128(
        (gv4i_p)gb, (lv4i_p)&Bs[trow * 32 + tcol], 16, 0);
#else
    uint4 av0 = reinterpret_cast<const uint4*>(ga)[0];
    uint4 av1 = reinterpret_cast<const uint4*>(ga)[1];
    uint4 bv0 = reinterpret_cast<const uint4*>(gb)[0];
    uint4 bv1 = reinterpret_cast<const uint4*>(gb)[1];
    *reinterpret_cast<uint4*>(&As[trow * 32 + tcol])     = av0;
    *reinterpret_cast<uint4*>(&As[trow * 32 + tcol + 8]) = av1;
    *reinterpret_cast<uint4*>(&Bs[trow * 32 + tcol])     = bv0;
    *reinterpret_cast<uint4*>(&Bs[trow * 32 + tcol + 8]) = bv1;
#endif

    if (k0 + 32 < K) {                       // pull next K tile toward caches
      __builtin_prefetch(ga + 32, 0, 1);
      __builtin_prefetch(gb + 32, 0, 1);
    }

#if USE_ASYNC_LDS
    __builtin_amdgcn_s_wait_asynccnt(0);
#endif
    __syncthreads();

    union { v16bf v; uint4 u[2]; } fa[2], fb[4];
#pragma unroll
    for (int mi = 0; mi < 2; mi++) {
      int m = wm * 32 + mi * 16 + col;
      fa[mi].u[0] = *reinterpret_cast<const uint4*>(&As[m * 32 + kb]);
      fa[mi].u[1] = *reinterpret_cast<const uint4*>(&As[m * 32 + kb + 16]);
    }
#pragma unroll
    for (int ni = 0; ni < 4; ni++) {
      int n = wn * 64 + ni * 16 + col;
      fb[ni].u[0] = *reinterpret_cast<const uint4*>(&Bs[n * 32 + kb]);
      fb[ni].u[1] = *reinterpret_cast<const uint4*>(&Bs[n * 32 + kb + 16]);
    }
#pragma unroll
    for (int mi = 0; mi < 2; mi++)
#pragma unroll
      for (int ni = 0; ni < 4; ni++)
        acc[mi][ni] = __builtin_amdgcn_wmma_f32_16x16x32_bf16(
            false, fa[mi].v, false, fb[ni].v, (short)0, acc[mi][ni],
            false, false);
    __syncthreads();
  }

#pragma unroll
  for (int mi = 0; mi < 2; mi++)
#pragma unroll
    for (int ni = 0; ni < 4; ni++)
#pragma unroll
      for (int i = 0; i < 8; i++) {
        int r = m0 + wm * 32 + mi * 16 + i + 8 * rhalf;  // C layout: VGPR i -> M=i / i+8
        int c = n0 + wn * 64 + ni * 16 + col;
        float val = acc[mi][ni][i];
        if (outF) outF[(size_t)r * N + c] = val;
        else      outB[(size_t)r * N + c] = (__bf16)val;
      }
}

// ---------------------------------------------------------------------------
// Sliding-window attention: one workgroup (4 waves) per (b, h, 128-query blk).
// V window (256x64) staged once in LDS transposed; each wave runs two 16-row
// strips sequentially so P-buffer stays per-wave.  LDS = 32KB + 32KB = 64KB.
// ---------------------------------------------------------------------------
__global__ __launch_bounds__(128) void swa_kernel(
    const __bf16* __restrict__ q, const __bf16* __restrict__ k,
    const __bf16* __restrict__ v, __bf16* __restrict__ attn) {
  __shared__ __bf16 Pl[4 * 16 * 256];   // per-wave prob strip, 32 KB
  __shared__ __bf16 Vt[64 * 256];       // V window transposed [d][tk], 32 KB

  const int tid   = threadIdx.x;
  const int lane  = tid & 31;
  const int col   = lane & 15;
  const int rhalf = lane >> 4;
  const int kb    = rhalf * 8;
  const int wave  = tid >> 5;           // 0..3

  const int blk = blockIdx.x;
  const int h   = blockIdx.y;
  const int b   = blockIdx.z;
  const int t0  = blk * BS;
  const int kw0 = t0 - BS;              // first key pos in window

  const size_t base = (size_t)b * TT * DD + (size_t)h * DKH;

  // ---- stage V window into LDS, transposed to [d][tk]
#pragma unroll
  for (int rr = 0; rr < 2; rr++) {
    int r  = tid + rr * 128;            // window row 0..255
    int tk = kw0 + r;
    union { uint4 u[8]; __bf16 h[64]; } vt;
    if (tk >= 0) {
      const __bf16* gv = v + base + (size_t)tk * DD;
#pragma unroll
      for (int c = 0; c < 8; c++) vt.u[c] = reinterpret_cast<const uint4*>(gv)[c];
    } else {
#pragma unroll
      for (int c = 0; c < 8; c++) vt.u[c] = make_uint4(0, 0, 0, 0);
    }
#pragma unroll
    for (int d = 0; d < 64; d++) Vt[d * 256 + r] = vt.h[d];
  }
  __syncthreads();

  const float scale = 0.125f;           // 1/sqrt(64)
  __bf16* Pw = &Pl[wave * 16 * 256];

  for (int s = 0; s < 2; s++) {
    const int r0 = s * 64 + wave * 16;  // strip base row within block

    // ---- S = Q K^T over the 256-key window: strip 16 x 256 (16 tiles)
    v8f sacc[16];
#pragma unroll
    for (int n = 0; n < 16; n++) sacc[n] = {};

#pragma unroll
    for (int ks = 0; ks < 2; ks++) {    // dk = 64 -> 2 WMMA k-steps
      union { v16bf v; uint4 u[2]; } fa;
      const __bf16* gq = q + base + (size_t)(t0 + r0 + col) * DD + ks * 32 + kb;
      fa.u[0] = *reinterpret_cast<const uint4*>(gq);
      fa.u[1] = *reinterpret_cast<const uint4*>(gq + 16);
#pragma unroll
      for (int n = 0; n < 16; n++) {
        union { v16bf v; uint4 u[2]; } fb;
        int tk = kw0 + n * 16 + col;
        if (tk >= 0) {
          const __bf16* gk = k + base + (size_t)tk * DD + ks * 32 + kb;
          fb.u[0] = *reinterpret_cast<const uint4*>(gk);
          fb.u[1] = *reinterpret_cast<const uint4*>(gk + 16);
        } else {
          fb.u[0] = make_uint4(0, 0, 0, 0);
          fb.u[1] = make_uint4(0, 0, 0, 0);
        }
        sacc[n] = __builtin_amdgcn_wmma_f32_16x16x32_bf16(
            false, fa.v, false, fb.v, (short)0, sacc[n], false, false);
      }
    }

    // ---- banded mask + row softmax (rows live in 16-lane halves)
#pragma unroll
    for (int i = 0; i < 8; i++) {
      int ri = r0 + i + 8 * rhalf;      // query row within block
      float mx = -3.0e38f;
#pragma unroll
      for (int n = 0; n < 16; n++) {
        int ci   = n * 16 + col;        // key col within window
        int kp   = kw0 + ci;
        int dist = BS + ri - ci;        // qpos - kpos
        bool ok  = (kp >= 0) && (dist >= 0) && (dist <= BS);
        float sv = sacc[n][i] * scale;
        sv = ok ? sv : -3.0e38f;
        sacc[n][i] = sv;
        mx = fmaxf(mx, sv);
      }
      mx = fmaxf(mx, __shfl_xor(mx, 1));
      mx = fmaxf(mx, __shfl_xor(mx, 2));
      mx = fmaxf(mx, __shfl_xor(mx, 4));
      mx = fmaxf(mx, __shfl_xor(mx, 8));
      float sum = 0.f;
#pragma unroll
      for (int n = 0; n < 16; n++) {
        float e = __expf(sacc[n][i] - mx);
        sacc[n][i] = e;
        sum += e;
      }
      sum += __shfl_xor(sum, 1);
      sum += __shfl_xor(sum, 2);
      sum += __shfl_xor(sum, 4);
      sum += __shfl_xor(sum, 8);
      float inv = 1.0f / sum;
#pragma unroll
      for (int n = 0; n < 16; n++)
        Pw[(i + 8 * rhalf) * 256 + n * 16 + col] = (__bf16)(sacc[n][i] * inv);
    }
    __syncthreads();

    // ---- O = P V : strip 16 x 64 (4 tiles), 8 k-steps over the 256 keys
    v8f oacc[4];
#pragma unroll
    for (int dt = 0; dt < 4; dt++) oacc[dt] = {};

#pragma unroll
    for (int kk = 0; kk < 8; kk++) {
      union { v16bf v; uint4 u[2]; } fp;
      const __bf16* pp = &Pw[col * 256 + kk * 32 + kb];
      fp.u[0] = *reinterpret_cast<const uint4*>(pp);
      fp.u[1] = *reinterpret_cast<const uint4*>(pp + 16);
#pragma unroll
      for (int dt = 0; dt < 4; dt++) {
        union { v16bf v; uint4 u[2]; } fv;
        const __bf16* vp = &Vt[(dt * 16 + col) * 256 + kk * 32 + kb];
        fv.u[0] = *reinterpret_cast<const uint4*>(vp);
        fv.u[1] = *reinterpret_cast<const uint4*>(vp + 16);
        oacc[dt] = __builtin_amdgcn_wmma_f32_16x16x32_bf16(
            false, fp.v, false, fv.v, (short)0, oacc[dt], false, false);
      }
    }

#pragma unroll
    for (int dt = 0; dt < 4; dt++)
#pragma unroll
      for (int i = 0; i < 8; i++) {
        int ri = r0 + i + 8 * rhalf;
        attn[base + (size_t)(t0 + ri) * DD + dt * 16 + col] = (__bf16)oacc[dt][i];
      }
    __syncthreads();                    // Pw reused next strip
  }
}

// ---------------------------------------------------------------------------
extern "C" void kernel_launch(void* const* d_in, const int* in_sizes, int n_in,
                              void* d_out, int out_size, void* d_ws,
                              size_t ws_size, hipStream_t stream) {
  const float* x  = (const float*)d_in[0];
  const float* Wq = (const float*)d_in[1];
  const float* Wk = (const float*)d_in[2];
  const float* Wv = (const float*)d_in[3];
  const float* Wo = (const float*)d_in[4];

  char* ws = (char*)d_ws;
  size_t off = 0;
  const size_t actBytes = (size_t)MTOT * DD * sizeof(__bf16);   // 32 MB
  const size_t wBytes   = (size_t)DD * DD * sizeof(__bf16);     //  8 MB

  __bf16* xb  = (__bf16*)(ws + off); off += actBytes;
  __bf16* qb  = (__bf16*)(ws + off); off += actBytes;
  __bf16* kbf = (__bf16*)(ws + off); off += actBytes;
  __bf16* vb  = (__bf16*)(ws + off); off += actBytes;
  __bf16* wqb = (__bf16*)(ws + off); off += wBytes;
  __bf16* wkb = (__bf16*)(ws + off); off += wBytes;
  __bf16* wvb = (__bf16*)(ws + off); off += wBytes;
  __bf16* wob = (__bf16*)(ws + off); off += wBytes;
  __bf16* attn = xb;   // xb is dead after the QKV GEMMs -> reuse

  // 1) casts (weights keep their native [N,K] layout: already WMMA-B friendly)
  int n4 = MTOT * DD / 4;
  cast_bf16_kernel<<<(n4 + 255) / 256, 256, 0, stream>>>(x, xb, n4);
  int w4 = DD * DD / 4;
  cast_bf16_kernel<<<(w4 + 255) / 256, 256, 0, stream>>>(Wq, wqb, w4);
  cast_bf16_kernel<<<(w4 + 255) / 256, 256, 0, stream>>>(Wk, wkb, w4);
  cast_bf16_kernel<<<(w4 + 255) / 256, 256, 0, stream>>>(Wv, wvb, w4);
  cast_bf16_kernel<<<(w4 + 255) / 256, 256, 0, stream>>>(Wo, wob, w4);

  // 2) QKV projections (bf16 out)
  dim3 gg(DD / 128, MTOT / 128);
  gemm_bf16_kernel<<<gg, 256, 0, stream>>>(xb, wqb, qb, nullptr, MTOT, DD, DD);
  gemm_bf16_kernel<<<gg, 256, 0, stream>>>(xb, wkb, kbf, nullptr, MTOT, DD, DD);
  gemm_bf16_kernel<<<gg, 256, 0, stream>>>(xb, wvb, vb, nullptr, MTOT, DD, DD);

  // 3) sliding-window attention
  dim3 ag(NB, HH, BB);
  swa_kernel<<<ag, 128, 0, stream>>>(qb, kbf, vb, attn);

  // 4) output projection (f32 out)
  gemm_bf16_kernel<<<gg, 256, 0, stream>>>(attn, wob, nullptr, (float*)d_out,
                                           MTOT, DD, DD);
}
